// SocialPooling_5360119185920
// MI455X (gfx1250) — compile-verified, matrix operand order checked by reading, not run on hardware
//
#include <hip/hip_runtime.h>
#include <hip/hip_bf16.h>
#include <stdint.h>

#define H_DIM      64
#define GRID_N     8
#define G2         64          // GRID_N*GRID_N
#define N_PED      64
#define N_SEQ      64
#define BATCH      4096        // N_SEQ*N_PED
#define IN_DIM     4096        // G2*H_DIM
#define BOT        1024
#define BN_EPS     1e-5f

typedef __bf16 bf16_t;
typedef __attribute__((ext_vector_type(16))) __bf16 v16bf;
typedef __attribute__((ext_vector_type(8)))  __bf16 v8bf;
typedef __attribute__((ext_vector_type(8)))  float  v8f;

static __device__ __forceinline__ unsigned short f32_to_bf16_raw(float f) {
    unsigned int u = __float_as_uint(f);
    // round-to-nearest-even
    unsigned int r = u + 0x7FFFu + ((u >> 16) & 1u);
    return (unsigned short)(r >> 16);
}

// ---------------------------------------------------------------------------
// Kernel 1: W (IN_DIM x BOT, f32 row-major) -> Wt (BOT x IN_DIM, bf16)
// so B-matrix WMMA fragments become contiguous 16B loads along K.
// ---------------------------------------------------------------------------
__global__ void convert_transpose_W(const float* __restrict__ W,
                                    unsigned short* __restrict__ Wt) {
    int idx = blockIdx.x * blockDim.x + threadIdx.x;   // over k*BOT + n (coalesced read)
    int k = idx >> 10;            // IN_DIM index
    int n = idx & (BOT - 1);      // BOT index
    Wt[(size_t)n * IN_DIM + k] = f32_to_bf16_raw(W[idx]);
}

// ---------------------------------------------------------------------------
// Kernel 2: social pooling. One block (64 threads) per pedestrian.
// LDS accumulator: 64 cells x 64 dims f32 = 16 KB. Thread = h index -> no races.
// Emits bf16 pool row (4096 elements, cell-major then h) for the GEMM A matrix.
// ---------------------------------------------------------------------------
__global__ void pool_kernel(const float* __restrict__ h_states,
                            const float* __restrict__ end_pos,
                            unsigned short* __restrict__ poolb) {
    __shared__ float acc[G2 * H_DIM];
    const int tid = threadIdx.x;        // 0..63 = h index
    const int row = blockIdx.x;         // global ped index 0..4095
    const int s = row >> 6;             // scene
    const int i = row & 63;             // ped within scene

    #pragma unroll
    for (int c = tid; c < G2 * H_DIM; c += 64) acc[c] = 0.0f;
    __syncthreads();

    const float half = 1.0f;            // NBR/2
    const float xi = end_pos[row * 2 + 0];
    const float yi = end_pos[row * 2 + 1];
    const float tlx = xi - half, tly = yi + half;
    const float brx = xi + half, bry = yi - half;

    const float* hid = h_states + (size_t)(s * N_PED) * H_DIM;
    const float* pos = end_pos + (size_t)(s * N_PED) * 2;

    for (int j = 0; j < N_PED; ++j) {
        if (j == i) continue;
        float px = pos[j * 2 + 0];
        float py = pos[j * 2 + 1];
        bool oob = (px >= brx) || (px <= tlx) || (py >= tly) || (py <= bry);
        if (oob) continue;
        int cx = (int)floorf((px - tlx) * 4.0f);   // GRID_N / NBR = 4
        int cy = (int)floorf((tly - py) * 4.0f);
        int cell = cx + cy * GRID_N;
        if (cell < 0 || cell >= G2) continue;
        acc[cell * H_DIM + tid] += hid[(size_t)j * H_DIM + tid];
    }
    __syncthreads();

    unsigned short* out = poolb + (size_t)row * IN_DIM;
    #pragma unroll 4
    for (int c = 0; c < G2; ++c)
        out[c * H_DIM + tid] = f32_to_bf16_raw(acc[c * H_DIM + tid]);
}

// ---------------------------------------------------------------------------
// Kernel 3: x = pool_h @ W + b via V_WMMA_F32_16X16X32_BF16, register-blocked.
// One wave computes a 64x32 output block = 4 (M) x 2 (N) tiles of 16x16.
// Per K-step: 4 A-fragments + 2 B-fragments (12 b128 loads) -> 8 WMMAs,
// i.e. 1.5 loads per WMMA instead of 4.
// Fragment layout (16-bit A, wave32): lanes 0-15 hold row M=lane with
// K = kk..kk+7 (v0-3) and kk+16..kk+23 (v4-7); lanes 16-31 hold row M=lane-16
// with K = kk+8..15 and kk+24..31. B is symmetric with N<->M, served from the
// K-contiguous transposed Wt.
// ---------------------------------------------------------------------------
__global__ void gemm_wmma_bf16(const unsigned short* __restrict__ Ab,
                               const unsigned short* __restrict__ Bb,
                               const float* __restrict__ bias,
                               float* __restrict__ X) {
    const int wave = threadIdx.x >> 5;
    const int lane = threadIdx.x & 31;
    const int tile = blockIdx.x * 8 + wave;     // 2048 wave-tiles total
    const int tm = tile >> 5;                   // 0..63   (64-row blocks)
    const int tn = tile & 31;                   // 0..31   (32-col blocks)
    const int m0 = tm * 64;
    const int n0 = tn * 32;
    const int lmod  = lane & 15;
    const int lhalf = lane >> 4;                // 0 or 1
    const int kofs  = lhalf * 8;

    const bf16_t* Abase = (const bf16_t*)Ab + (size_t)(m0 + lmod) * IN_DIM + kofs;
    const bf16_t* Bbase = (const bf16_t*)Bb + (size_t)(n0 + lmod) * IN_DIM + kofs;

    v8f acc[4][2] = {};

    for (int kk = 0; kk < IN_DIM; kk += 32) {
        // pull the next K-slab toward the near caches (global_prefetch_b8)
        __builtin_prefetch(Abase + kk + 32, 0, 3);
        __builtin_prefetch(Bbase + kk + 32, 0, 3);

        v16bf Af[4], Bf[2];
        #pragma unroll
        for (int m = 0; m < 4; ++m) {
            v8bf lo = *(const v8bf*)(Abase + (size_t)m * 16 * IN_DIM + kk);
            v8bf hi = *(const v8bf*)(Abase + (size_t)m * 16 * IN_DIM + kk + 16);
            #pragma unroll
            for (int t = 0; t < 8; ++t) { Af[m][t] = lo[t]; Af[m][t + 8] = hi[t]; }
        }
        #pragma unroll
        for (int n = 0; n < 2; ++n) {
            v8bf lo = *(const v8bf*)(Bbase + (size_t)n * 16 * IN_DIM + kk);
            v8bf hi = *(const v8bf*)(Bbase + (size_t)n * 16 * IN_DIM + kk + 16);
            #pragma unroll
            for (int t = 0; t < 8; ++t) { Bf[n][t] = lo[t]; Bf[n][t + 8] = hi[t]; }
        }

        #pragma unroll
        for (int m = 0; m < 4; ++m) {
            #pragma unroll
            for (int n = 0; n < 2; ++n) {
                acc[m][n] = __builtin_amdgcn_wmma_f32_16x16x32_bf16(
                        /*neg_a=*/false, Af[m], /*neg_b=*/false, Bf[n],
                        /*c_mod=*/(short)0, acc[m][n],
                        /*reuse_a=*/false, /*reuse_b=*/false);
            }
        }
    }

    // D layout: lanes 0-15 -> N=n0+lane, M=mtile+v; lanes 16-31 -> N=n0+lane-16,
    // M=mtile+8+v. Add bias (per-column) in the epilogue.
    const int mbase = m0 + lhalf * 8;
    #pragma unroll
    for (int n = 0; n < 2; ++n) {
        const int ncol = n0 + n * 16 + lmod;
        const float bb = bias[ncol];
        #pragma unroll
        for (int m = 0; m < 4; ++m) {
            #pragma unroll
            for (int v = 0; v < 8; ++v)
                X[(size_t)(mbase + m * 16 + v) * BOT + ncol] = acc[m][n][v] + bb;
        }
    }
}

// ---------------------------------------------------------------------------
// Kernel 4: per-column mean / rstd over BATCH rows (thread = column, coalesced)
// ---------------------------------------------------------------------------
__global__ void bn_stats(const float* __restrict__ X,
                         float* __restrict__ meanv,
                         float* __restrict__ rstdv) {
    int col = blockIdx.x * blockDim.x + threadIdx.x;   // 0..1023
    float s = 0.0f, sq = 0.0f;
    #pragma unroll 4
    for (int r = 0; r < BATCH; ++r) {
        float v = X[(size_t)r * BOT + col];
        s += v;
        sq += v * v;
    }
    float m   = s * (1.0f / BATCH);
    float var = sq * (1.0f / BATCH) - m * m;
    meanv[col] = m;
    rstdv[col] = rsqrtf(var + BN_EPS);
}

// ---------------------------------------------------------------------------
// Kernel 5: normalize + gamma/beta + ReLU -> d_out (f32)
// ---------------------------------------------------------------------------
__global__ void bn_apply(const float* __restrict__ X,
                         const float* __restrict__ meanv,
                         const float* __restrict__ rstdv,
                         const float* __restrict__ gamma,
                         const float* __restrict__ beta,
                         float* __restrict__ out) {
    size_t idx = (size_t)blockIdx.x * blockDim.x + threadIdx.x;
    int col = (int)(idx & (BOT - 1));
    float v = (X[idx] - meanv[col]) * rstdv[col] * gamma[col] + beta[col];
    out[idx] = v > 0.0f ? v : 0.0f;
}

// ---------------------------------------------------------------------------
extern "C" void kernel_launch(void* const* d_in, const int* in_sizes, int n_in,
                              void* d_out, int out_size, void* d_ws, size_t ws_size,
                              hipStream_t stream) {
    const float* h_states = (const float*)d_in[0];   // (1, 4096, 64)
    // d_in[1] = seq_start_end (int32) -- scenes are contiguous 64-ped blocks, unused
    const float* end_pos  = (const float*)d_in[2];   // (4096, 2)
    // d_in[3] = rel_pos -- unused by reference math
    const float* W        = (const float*)d_in[4];   // (4096, 1024)
    const float* b        = (const float*)d_in[5];   // (1024,)
    const float* gamma    = (const float*)d_in[6];   // (1024,)
    const float* beta     = (const float*)d_in[7];   // (1024,)
    float* out            = (float*)d_out;           // (4096, 1024)

    // Workspace carve-up (all 256B-aligned chunks)
    uintptr_t base = (uintptr_t)d_ws;
    unsigned short* Wt    = (unsigned short*)base;                              // 8 MB
    unsigned short* poolb = (unsigned short*)(base + (size_t)8  * 1024 * 1024); // 32 MB
    float* X              = (float*)        (base + (size_t)40 * 1024 * 1024);  // 16 MB
    float* meanv          = (float*)        (base + (size_t)56 * 1024 * 1024);  // 4 KB
    float* rstdv          = meanv + BOT;                                        // 4 KB

    // 1) W f32 -> Wt bf16 transposed (K-contiguous rows)
    convert_transpose_W<<<(IN_DIM * BOT) / 256, 256, 0, stream>>>(W, Wt);

    // 2) social pooling -> bf16 A matrix
    pool_kernel<<<BATCH, 64, 0, stream>>>(h_states, end_pos, poolb);

    // 3) register-blocked bf16 WMMA GEMM + bias:
    //    2048 wave-tiles (64x32 each) / 8 waves per block = 256 blocks
    gemm_wmma_bf16<<<256, 256, 0, stream>>>(poolb, Wt, b, X);

    // 4) batch-norm statistics (per column)
    bn_stats<<<BOT / 256, 256, 0, stream>>>(X, meanv, rstdv);

    // 5) normalize + affine + ReLU
    bn_apply<<<(BATCH * BOT) / 256, 256, 0, stream>>>(X, meanv, rstdv, gamma, beta, out);
}
